// HeteroRelEventSAGE_15590731284983
// MI455X (gfx1250) — matrix operand order repaired and checked by reading, not compiled, not run on hardware
//
#include <hip/hip_runtime.h>
#include <math.h>

// ---------------- types ----------------
typedef __bf16 bf16_t;
typedef bf16_t bf16x16 __attribute__((ext_vector_type(16)));
typedef float  f32x8   __attribute__((ext_vector_type(8)));

union AB32 { bf16x16 v; uint4 q[2]; };
union C8   { f32x8 v;   float f[8]; };

#define NB      16384    // B
#define FANOUT  16
#define EMB     128
#define HID     256
#define KEV     512      // src(128) + dst(128) + h1(256)
#define KFIN    384      // self(128) + neigh(256)
#define APAD    8        // bf16 row padding (16B) for the A tile
#define BSTRIDE 40       // 32 K-elements + 8 pad (80B rows -> conflict-free-ish b128 reads)
#define BSLICE  (256 * BSTRIDE)   // shorts per B slice buffer

__device__ __forceinline__ unsigned short f2bf(float f) {
  union { float f; unsigned int u; } v; v.f = f;
  unsigned int u = v.u;
  unsigned int r = u + 0x7FFFu + ((u >> 16) & 1u);   // round-to-nearest-even
  return (unsigned short)(r >> 16);
}

// ---------------- prep: stack + convert weights to bf16 ----------------
// Wev[n][k]  = concat(event_src_lin_w[n], event_dst_lin_w[n], mlp_w2[n])   [256][512]
// Wfin[n][k] = concat(src_self_w[n], src_neigh_w[n])                        [256][384]
__global__ void prep_weights(const float* __restrict__ esw, const float* __restrict__ edw,
                             const float* __restrict__ w2,  const float* __restrict__ ssw,
                             const float* __restrict__ snw,
                             unsigned short* __restrict__ Wev,
                             unsigned short* __restrict__ Wfin) {
  int idx = blockIdx.x * 256 + threadIdx.x;
  if (idx < HID * KEV) {
    int n = idx / KEV, k = idx % KEV;
    float v = (k < 128) ? esw[n * 128 + k]
            : (k < 256) ? edw[n * 128 + (k - 128)]
                        : w2 [n * 256 + (k - 256)];
    Wev[n * KEV + k] = f2bf(v);
  } else {
    int i2 = idx - HID * KEV;
    if (i2 < HID * KFIN) {
      int n = i2 / KFIN, k = i2 % KFIN;
      float v = (k < 128) ? ssw[n * 128 + k] : snw[n * 256 + (k - 128)];
      Wfin[n * KFIN + k] = f2bf(v);
    }
  }
}

// relT[r][n] = edge_emb[r] . edge_lin_w[n] + mlp_b2[n]   (16 x 256 LUT, b2 folded in)
__global__ void prep_rel(const float* __restrict__ edge_emb,
                         const float* __restrict__ edge_lin_w,
                         const float* __restrict__ b2,
                         float* __restrict__ relT) {
  int r = blockIdx.x, n = threadIdx.x;
  const float* e = edge_emb + r * EMB;
  const float* w = edge_lin_w + n * EMB;
  float s = b2[n];
  for (int k = 0; k < EMB; ++k) s += e[k] * w[k];
  relT[r * HID + n] = s;
}

// ---------------- event kernel: [BE,512]x[512,256] bf16 WMMA + fused mean ----------------
// block: 256 threads (8 waves), 64 consecutive events == 4 graph nodes
// wave w: m_tile = w&3 (16 events == one node), n_half = w>>2 (128 cols, 8 n-tiles)
// B is staged per-block in LDS as double-buffered 32-K slices (cuts L2 re-reads 4x).
__global__ __launch_bounds__(256) void event_kernel(
    const int* __restrict__ nbr_ev,   const int* __restrict__ ev_src_type,
    const int* __restrict__ ev_dst_type, const int* __restrict__ ev_edge_type,
    const int* __restrict__ ev_src_id, const int* __restrict__ ev_dst_id,
    const int* __restrict__ ev_ts_s,  const float* __restrict__ ev_w,
    const float* __restrict__ emb0,   const float* __restrict__ emb1,
    const float* __restrict__ mlp_w1, const float* __restrict__ mlp_b1,
    const unsigned short* __restrict__ Wev, const float* __restrict__ relTg,
    float* __restrict__ neigh) {
  __shared__ unsigned short Abuf[64 * (KEV + APAD)];   // 66,560 B
  __shared__ unsigned short Bbuf[2 * BSLICE];          // 40,960 B (double-buffered B slice)
  __shared__ float relT[16 * HID];                     // 16,384 B
  __shared__ const float* sptr[64];
  __shared__ const float* dptr[64];
  __shared__ float featx[64], featy[64], vmask[64];
  __shared__ int   relid[64];

  const int tid = threadIdx.x;
  const int e0  = blockIdx.x * 64;

  if (tid < 64) {
    int e     = e0 + tid;
    int ev    = nbr_ev[e];
    int valid = (ev >= 0);
    int evs   = valid ? ev : 0;
    int st = ev_src_type[evs], dt = ev_dst_type[evs];
    size_t sid = (size_t)ev_src_id[evs], did = (size_t)ev_dst_id[evs];
    sptr[tid]  = (st == 0 ? emb0 : emb1) + sid * EMB;
    dptr[tid]  = (dt == 0 ? emb0 : emb1) + did * EMB;
    relid[tid] = ev_edge_type[evs];
    vmask[tid] = valid ? (1.0f / 16.0f) : 0.0f;        // mean/FANOUT folded in
    featx[tid] = (float)ev_ts_s[evs] * 1.0e-6f;        // ts_norm
    featy[tid] = log1pf(ev_w[evs]);
  }
  __syncthreads();

  // start fetching B slice 0 into registers (overlaps with A staging below)
  uint4 stage[4];
  #pragma unroll
  for (int c = 0; c < 4; ++c) {
    int chunk = c * 256 + tid;                // 1024 chunks of 16B cover [256][32] slice
    int n = chunk >> 2, part = chunk & 3;
    stage[c] = *(const uint4*)(Wev + (size_t)n * KEV + part * 8);
  }

  // gather src/dst embeddings -> bf16 A cols [0,256)
  for (int it = 0; it < 16; ++it) {
    int idx = it * 256 + tid;                 // 0..4095 : 64 events x 64 float4
    int e = idx >> 6, q = idx & 63;
    const float* p = (q < 32) ? (sptr[e] + q * 4) : (dptr[e] + (q - 32) * 4);
    float4 v = *(const float4*)p;
    int col = (q < 32) ? (q * 4) : (128 + (q - 32) * 4);
    unsigned int lo = (unsigned int)f2bf(v.x) | ((unsigned int)f2bf(v.y) << 16);
    unsigned int hi = (unsigned int)f2bf(v.z) | ((unsigned int)f2bf(v.w) << 16);
    *(uint2*)&Abuf[e * (KEV + APAD) + col] = make_uint2(lo, hi);
  }
  // h1 = relu(feat @ w1^T + b1) -> A cols [256,512)
  for (int it = 0; it < 64; ++it) {
    int idx = it * 256 + tid;                 // 0..16383 : 64 events x 256
    int e = idx >> 8, j = idx & 255;
    float h = fmaf(featx[e], mlp_w1[2 * j], fmaf(featy[e], mlp_w1[2 * j + 1], mlp_b1[j]));
    Abuf[e * (KEV + APAD) + 256 + j] = f2bf(fmaxf(h, 0.0f));
  }
  // rel LUT -> LDS
  for (int it = 0; it < 4; ++it) {
    int idx = it * 256 + tid;                 // 1024 float4
    ((float4*)relT)[idx] = ((const float4*)relTg)[idx];
  }
  // commit B slice 0 to LDS
  #pragma unroll
  for (int c = 0; c < 4; ++c) {
    int chunk = c * 256 + tid;
    int n = chunk >> 2, part = chunk & 3;
    *(uint4*)&Bbuf[n * BSTRIDE + part * 8] = stage[c];
  }
  __syncthreads();

  const int lane = tid & 31;
  const int wave = tid >> 5;
  const int rowbase = (wave & 3) * 16;
  const int nbase0  = (wave >> 2) * 128;
  const int ln = lane & 15;
  const int hi = lane >> 4;

  C8 acc[8];
  #pragma unroll
  for (int t = 0; t < 8; ++t)
    #pragma unroll
    for (int j = 0; j < 8; ++j) acc[t].f[j] = 0.0f;

  const unsigned short* ArowBase = &Abuf[(rowbase + ln) * (KEV + APAD)];
  for (int ks = 0; ks < KEV / 32; ++ks) {
    // prefetch next B slice into registers while this slice computes
    if (ks + 1 < KEV / 32) {
      #pragma unroll
      for (int c = 0; c < 4; ++c) {
        int chunk = c * 256 + tid;
        int n = chunk >> 2, part = chunk & 3;
        stage[c] = *(const uint4*)(Wev + (size_t)n * KEV + (ks + 1) * 32 + part * 8);
      }
    }
    const int k0 = ks * 32;
    AB32 a;                                   // 16-bit A 16x32 frag: lanes<16 K=k0..k0+15, lanes>=16 K=k0+16..k0+31
    a.q[0] = *(const uint4*)(ArowBase + k0 + hi * 16);
    a.q[1] = *(const uint4*)(ArowBase + k0 + hi * 16 + 8);
    const unsigned short* Bcur = &Bbuf[(ks & 1) * BSLICE];
    #pragma unroll
    for (int t = 0; t < 8; ++t) {
      const unsigned short* bp = Bcur + (nbase0 + t * 16 + ln) * BSTRIDE + hi * 16;
      AB32 b;
      b.q[0] = *(const uint4*)(bp);
      b.q[1] = *(const uint4*)(bp + 8);
      acc[t].v = __builtin_amdgcn_wmma_f32_16x16x32_bf16(
          false, a.v, false, b.v, (short)0, acc[t].v, false, false);
    }
    // commit next slice to the other buffer
    if (ks + 1 < KEV / 32) {
      unsigned short* Bnxt = &Bbuf[((ks + 1) & 1) * BSLICE];
      #pragma unroll
      for (int c = 0; c < 4; ++c) {
        int chunk = c * 256 + tid;
        int n = chunk >> 2, part = chunk & 3;
        *(uint4*)&Bnxt[n * BSTRIDE + part * 8] = stage[c];
      }
    }
    __syncthreads();
  }

  // epilogue: + relLUT, relu, mask(valid/16), sum over 16 events (one node) -> neigh
  const int node = blockIdx.x * 4 + (wave & 3);
  #pragma unroll
  for (int t = 0; t < 8; ++t) {
    const int n = nbase0 + t * 16 + ln;
    float s = 0.0f;
    #pragma unroll
    for (int j = 0; j < 8; ++j) {
      const int m = rowbase + hi * 8 + j;     // C frag: lanes<16 M=j, lanes>=16 M=j+8
      float v = acc[t].f[j] + relT[relid[m] * HID + n];
      s += fmaxf(v, 0.0f) * vmask[m];
    }
    s += __shfl_xor(s, 16, 32);               // combine M 0..7 with 8..15
    if (hi == 0) neigh[(size_t)node * HID + n] = s;
  }
}

// ---------------- final kernel: out = relu([self||neigh] @ Wfin^T), [B,384]x[384,256] ----------------
__global__ __launch_bounds__(128) void final_kernel(
    const int* __restrict__ node_ids, const float* __restrict__ emb0,
    const float* __restrict__ neigh,  const unsigned short* __restrict__ Wfin,
    float* __restrict__ out) {
  __shared__ unsigned short Abuf[32 * (KFIN + APAD)];  // 25,088 B
  __shared__ const float* sptr[32];

  const int tid = threadIdx.x;
  const int b0  = blockIdx.x * 32;

  if (tid < 32) sptr[tid] = emb0 + (size_t)node_ids[b0 + tid] * EMB;
  __syncthreads();

  // self embedding -> cols [0,128)
  for (int it = 0; it < 8; ++it) {
    int idx = it * 128 + tid;                 // 32 nodes x 32 float4
    int e = idx >> 5, q = idx & 31;
    float4 v = *(const float4*)(sptr[e] + q * 4);
    unsigned int lo = (unsigned int)f2bf(v.x) | ((unsigned int)f2bf(v.y) << 16);
    unsigned int hi = (unsigned int)f2bf(v.z) | ((unsigned int)f2bf(v.w) << 16);
    *(uint2*)&Abuf[e * (KFIN + APAD) + q * 4] = make_uint2(lo, hi);
  }
  // neigh -> cols [128,384)
  for (int it = 0; it < 16; ++it) {
    int idx = it * 128 + tid;                 // 32 nodes x 64 float4
    int e = idx >> 6, q = idx & 63;
    float4 v = *(const float4*)(neigh + (size_t)(b0 + e) * HID + q * 4);
    unsigned int lo = (unsigned int)f2bf(v.x) | ((unsigned int)f2bf(v.y) << 16);
    unsigned int hi = (unsigned int)f2bf(v.z) | ((unsigned int)f2bf(v.w) << 16);
    *(uint2*)&Abuf[e * (KFIN + APAD) + 128 + q * 4] = make_uint2(lo, hi);
  }
  __syncthreads();

  const int lane = tid & 31;
  const int wave = tid >> 5;
  const int rowbase = (wave & 1) * 16;
  const int nbase0  = (wave >> 1) * 128;
  const int ln = lane & 15;
  const int hi = lane >> 4;

  C8 acc[8];
  #pragma unroll
  for (int t = 0; t < 8; ++t)
    #pragma unroll
    for (int j = 0; j < 8; ++j) acc[t].f[j] = 0.0f;

  const unsigned short* ArowBase = &Abuf[(rowbase + ln) * (KFIN + APAD)];
  for (int ks = 0; ks < KFIN / 32; ++ks) {
    const int k0 = ks * 32;
    AB32 a;
    a.q[0] = *(const uint4*)(ArowBase + k0 + hi * 16);
    a.q[1] = *(const uint4*)(ArowBase + k0 + hi * 16 + 8);
    #pragma unroll
    for (int t = 0; t < 8; ++t) {
      const int n = nbase0 + t * 16 + ln;
      const unsigned short* bp = Wfin + (size_t)n * KFIN + k0 + hi * 16;
      AB32 b;
      b.q[0] = *(const uint4*)(bp);
      b.q[1] = *(const uint4*)(bp + 8);
      acc[t].v = __builtin_amdgcn_wmma_f32_16x16x32_bf16(
          false, a.v, false, b.v, (short)0, acc[t].v, false, false);
    }
  }

  #pragma unroll
  for (int t = 0; t < 8; ++t) {
    const int n = nbase0 + t * 16 + ln;
    #pragma unroll
    for (int j = 0; j < 8; ++j) {
      const int m = rowbase + hi * 8 + j;
      out[(size_t)(b0 + m) * HID + n] = fmaxf(acc[t].f[j], 0.0f);
    }
  }
}

// ---------------- launch ----------------
extern "C" void kernel_launch(void* const* d_in, const int* in_sizes, int n_in,
                              void* d_out, int out_size, void* d_ws, size_t ws_size,
                              hipStream_t stream) {
  const int*   node_ids     = (const int*)  d_in[0];
  const int*   nbr_ev       = (const int*)  d_in[1];
  const int*   ev_src_type  = (const int*)  d_in[2];
  const int*   ev_dst_type  = (const int*)  d_in[3];
  const int*   ev_edge_type = (const int*)  d_in[4];
  const int*   ev_src_id    = (const int*)  d_in[5];
  const int*   ev_dst_id    = (const int*)  d_in[6];
  const int*   ev_ts_s      = (const int*)  d_in[7];
  const float* ev_w         = (const float*)d_in[8];
  const float* emb0         = (const float*)d_in[9];
  const float* emb1         = (const float*)d_in[10];
  const float* edge_emb     = (const float*)d_in[11];
  const float* edge_lin_w   = (const float*)d_in[12];
  const float* mlp_w1       = (const float*)d_in[13];
  const float* mlp_b1       = (const float*)d_in[14];
  const float* mlp_w2       = (const float*)d_in[15];
  const float* mlp_b2       = (const float*)d_in[16];
  const float* esw          = (const float*)d_in[17];
  const float* edw          = (const float*)d_in[18];
  const float* ssw          = (const float*)d_in[19];
  const float* snw          = (const float*)d_in[20];

  // workspace layout (bytes)
  char* ws = (char*)d_ws;
  unsigned short* Wev  = (unsigned short*)(ws + 0);        // 256*512*2  = 262,144
  unsigned short* Wfin = (unsigned short*)(ws + 262144);   // 256*384*2  = 196,608
  float*          relT = (float*)         (ws + 458752);   // 16*256*4   =  16,384
  float*          neigh= (float*)         (ws + 475136);   // 16384*256*4= 16,777,216

  prep_weights<<<(HID * KEV + HID * KFIN) / 256, 256, 0, stream>>>(esw, edw, mlp_w2, ssw, snw, Wev, Wfin);
  prep_rel<<<16, HID, 0, stream>>>(edge_emb, edge_lin_w, mlp_b2, relT);
  event_kernel<<<(NB * FANOUT) / 64, 256, 0, stream>>>(
      nbr_ev, ev_src_type, ev_dst_type, ev_edge_type, ev_src_id, ev_dst_id,
      ev_ts_s, ev_w, emb0, emb1, mlp_w1, mlp_b1, Wev, relT, neigh);
  final_kernel<<<NB / 32, 128, 0, stream>>>(node_ids, emb0, neigh, Wfin, (float*)d_out);
}
